// LSTM_DDPGActorNet_91104846282802
// MI455X (gfx1250) — matrix operand order repaired and checked by reading, not compiled, not run on hardware
//
#include <hip/hip_runtime.h>
#include <hip/hip_bf16.h>
#include <math.h>

// ---------------------------------------------------------------------------
// Types for CDNA5 WMMA (wave32): V_WMMA_F32_16X16X32_BF16
// ---------------------------------------------------------------------------
typedef __bf16 v16bf __attribute__((ext_vector_type(16)));
typedef __bf16 bf8   __attribute__((ext_vector_type(8)));
typedef float  v8f   __attribute__((ext_vector_type(8)));

#define B_BATCH 4096
#define T_SEQ   128
#define HID     64
#define G4      256   // 4*H gates
#define FCH     256
#define OUTD    8

__device__ __forceinline__ v8f wmma_bf16(v16bf a, v16bf b, v8f c) {
  // (neg_a, A, neg_b, B, c_mod, C, reuse_a, reuse_b)
  return __builtin_amdgcn_wmma_f32_16x16x32_bf16(false, a, false, b, (short)0, c,
                                                 false, false);
}

// A-matrix fragment (16x32 bf16, MxK). Per ISA layout:
//   lane l: row = l%16; halves 0..7 -> K = k0 + 8*(l/16) .. +7 (contig 16B)
//           halves 8..15 -> K = k0 + 16 + 8*(l/16) .. +7 (contig 16B)
__device__ __forceinline__ v16bf frag_a(const __bf16* base, int ldk, int k0, int lane) {
  const __bf16* p = base + (size_t)(lane & 15) * ldk + k0 + ((lane >> 4) << 3);
  v16bf a;
  ((bf8*)&a)[0] = *(const bf8*)p;
  ((bf8*)&a)[1] = *(const bf8*)(p + 16);
  return a;
}

// B-matrix fragment (32x16 bf16, KxN) from row-major weights W[n][k]:
//   lane l: col n = n0 + l%16; halves 0..7 -> K = k0 + 8*(l/16) .. +7,
//           halves 8..15 -> K = k0 + 16 + 8*(l/16) .. +7
__device__ __forceinline__ v16bf frag_b(const __bf16* w, int ldk, int n0, int k0, int lane) {
  const __bf16* p = w + (size_t)(n0 + (lane & 15)) * ldk + k0 + ((lane >> 4) << 3);
  v16bf b;
  ((bf8*)&b)[0] = *(const bf8*)p;
  ((bf8*)&b)[1] = *(const bf8*)(p + 16);
  return b;
}

// Branch-free activations: v_exp_f32 + v_rcp_f32, no EXEC-mask divergence.
__device__ __forceinline__ float sigm(float x) {
  return __builtin_amdgcn_rcpf(1.0f + __expf(-x));
}
__device__ __forceinline__ float tanh_fast(float x) {
  // tanh(x) = 2*sigmoid(2x) - 1
  return 2.0f * __builtin_amdgcn_rcpf(1.0f + __expf(-2.0f * x)) - 1.0f;
}

// ---------------------------------------------------------------------------
// Weight prep: f32 -> bf16, bias fusion
// ---------------------------------------------------------------------------
__global__ void cvt_bf16_kernel(const float* __restrict__ in, __bf16* __restrict__ out, int n) {
  int i = blockIdx.x * blockDim.x + threadIdx.x;
  if (i < n) out[i] = (__bf16)in[i];
}

__global__ void bias_add_kernel(const float* __restrict__ a, const float* __restrict__ b,
                                float* __restrict__ out, int n) {
  int i = blockIdx.x * blockDim.x + threadIdx.x;
  if (i < n) out[i] = a[i] + b[i];
}

// ---------------------------------------------------------------------------
// LSTM layer 0 (input size 1). One wave owns 16 batch rows; c-state in VGPRs,
// h staged per-wave in LDS (row-major [16][64] bf16 = A-fragment source).
// Block = 128 threads (4 waves), grid = B/64.
// ---------------------------------------------------------------------------
__global__ __launch_bounds__(128) void lstm0_kernel(
    const float* __restrict__ obs,    // [B,T]
    const float* __restrict__ wih0,   // [256] (input size 1)
    const float* __restrict__ bias0,  // [256] = bih0+bhh0
    const __bf16* __restrict__ whh0,  // [256][64] bf16 row-major
    __bf16* __restrict__ x1)          // out: [B,T,64] bf16
{
  __shared__ __bf16 sWhh[G4 * HID];       // 32 KB
  __shared__ float  sWih[G4];
  __shared__ float  sBias[G4];
  __shared__ __bf16 sH[4][16 * HID];      // 8 KB, per-wave h_{t-1}

  const int tid  = threadIdx.x;
  const int lane = tid & 31;
  const int wv   = tid >> 5;
  const int b0   = blockIdx.x * 64 + wv * 16;
  const int rsub = (lane >> 4) << 3;      // 0 or 8
  const int nlan = lane & 15;

  for (int i = tid; i < G4 * HID; i += 128) sWhh[i] = whh0[i];
  for (int i = tid; i < G4; i += 128) { sWih[i] = wih0[i]; sBias[i] = bias0[i]; }
  for (int i = tid; i < 4 * 16 * HID; i += 128) (&sH[0][0])[i] = (__bf16)0.0f;
  __syncthreads();

  float cst[4][8];
#pragma unroll
  for (int j = 0; j < 4; ++j)
#pragma unroll
    for (int v = 0; v < 8; ++v) cst[j][v] = 0.0f;

  for (int t = 0; t < T_SEQ; ++t) {
    // h_{t-1} A-fragments (K = 0..31, 32..63)
    v16bf a0 = frag_a(sH[wv], HID, 0, lane);
    v16bf a1 = frag_a(sH[wv], HID, 32, lane);

    float ov[8];
#pragma unroll
    for (int v = 0; v < 8; ++v)
      ov[v] = obs[(size_t)(b0 + rsub + v) * T_SEQ + t];

#pragma unroll
    for (int j = 0; j < 4; ++j) {          // 16-col groups of the hidden dim
      v8f acc[4];
#pragma unroll
      for (int q = 0; q < 4; ++q) {        // gates i,f,g,o at N-offset 64*q
        const int n0  = q * HID + j * 16;
        const int col = n0 + nlan;
        const float wx = sWih[col];
        const float bb = sBias[col];
        v8f a;
#pragma unroll
        for (int v = 0; v < 8; ++v) a[v] = ov[v] * wx + bb;  // xg_t + bias
        a = wmma_bf16(a0, frag_b(sWhh, HID, n0, 0, lane), a);
        a = wmma_bf16(a1, frag_b(sWhh, HID, n0, 32, lane), a);
        acc[q] = a;
      }
#pragma unroll
      for (int v = 0; v < 8; ++v) {
        float ig = sigm(acc[0][v]);
        float fg = sigm(acc[1][v]);
        float gg = tanh_fast(acc[2][v]);
        float og = sigm(acc[3][v]);
        float c  = fg * cst[j][v] + ig * gg;
        float h  = og * tanh_fast(c);
        cst[j][v] = c;
        __bf16 hb = (__bf16)h;
        const int hcol = j * 16 + nlan;
        sH[wv][(rsub + v) * HID + hcol] = hb;                        // next-step A
        x1[((size_t)(b0 + rsub + v) * T_SEQ + t) * HID + hcol] = hb; // layer-1 input
      }
    }
  }
}

// ---------------------------------------------------------------------------
// LSTM layer 1 (input size 64): input GEMM computed on the fly per step
// (x_t @ Wih1^T) + recurrent (h @ Whh1^T), both via WMMA.
// ---------------------------------------------------------------------------
__global__ __launch_bounds__(128) void lstm1_kernel(
    const __bf16* __restrict__ xin,   // [B,T,64] bf16 (layer-0 h)
    const float* __restrict__ bias1,  // [256]
    const __bf16* __restrict__ wih1,  // [256][64] bf16 (global, L0/L2 resident)
    const __bf16* __restrict__ whh1,  // [256][64] bf16 -> LDS
    __bf16* __restrict__ xout)        // [B,T,64] bf16 (layer-1 h)
{
  __shared__ __bf16 sWhh[G4 * HID];
  __shared__ float  sBias[G4];
  __shared__ __bf16 sH[4][16 * HID];

  const int tid  = threadIdx.x;
  const int lane = tid & 31;
  const int wv   = tid >> 5;
  const int b0   = blockIdx.x * 64 + wv * 16;
  const int rsub = (lane >> 4) << 3;
  const int nlan = lane & 15;

  for (int i = tid; i < G4 * HID; i += 128) sWhh[i] = whh1[i];
  for (int i = tid; i < G4; i += 128) sBias[i] = bias1[i];
  for (int i = tid; i < 4 * 16 * HID; i += 128) (&sH[0][0])[i] = (__bf16)0.0f;
  __syncthreads();

  float cst[4][8];
#pragma unroll
  for (int j = 0; j < 4; ++j)
#pragma unroll
    for (int v = 0; v < 8; ++v) cst[j][v] = 0.0f;

  for (int t = 0; t < T_SEQ; ++t) {
    v16bf ah0 = frag_a(sH[wv], HID, 0, lane);
    v16bf ah1 = frag_a(sH[wv], HID, 32, lane);
    const __bf16* xbase = xin + ((size_t)b0 * T_SEQ + t) * HID;
    v16bf ax0 = frag_a(xbase, T_SEQ * HID, 0, lane);
    v16bf ax1 = frag_a(xbase, T_SEQ * HID, 32, lane);

#pragma unroll
    for (int j = 0; j < 4; ++j) {
      v8f acc[4];
#pragma unroll
      for (int q = 0; q < 4; ++q) {
        const int n0  = q * HID + j * 16;
        const int col = n0 + nlan;
        const float bb = sBias[col];
        v8f a;
#pragma unroll
        for (int v = 0; v < 8; ++v) a[v] = bb;
        a = wmma_bf16(ax0, frag_b(wih1, HID, n0, 0, lane), a);   // input contrib
        a = wmma_bf16(ax1, frag_b(wih1, HID, n0, 32, lane), a);
        a = wmma_bf16(ah0, frag_b(sWhh, HID, n0, 0, lane), a);   // recurrent
        a = wmma_bf16(ah1, frag_b(sWhh, HID, n0, 32, lane), a);
        acc[q] = a;
      }
#pragma unroll
      for (int v = 0; v < 8; ++v) {
        float ig = sigm(acc[0][v]);
        float fg = sigm(acc[1][v]);
        float gg = tanh_fast(acc[2][v]);
        float og = sigm(acc[3][v]);
        float c  = fg * cst[j][v] + ig * gg;
        float h  = og * tanh_fast(c);
        cst[j][v] = c;
        __bf16 hb = (__bf16)h;
        const int hcol = j * 16 + nlan;
        sH[wv][(rsub + v) * HID + hcol] = hb;
        xout[((size_t)(b0 + rsub + v) * T_SEQ + t) * HID + hcol] = hb;
      }
    }
  }
}

// ---------------------------------------------------------------------------
// FC GEMM: out[M][256] = act(A[M][K] @ W[256][K]^T + bias). Wave = 16 rows x
// 64 cols (4 N-tiles); block = 256 threads (8 waves) = 128 rows.
// ---------------------------------------------------------------------------
__global__ __launch_bounds__(256) void fc_gemm_kernel(
    const __bf16* __restrict__ A,     // [M][K]
    const __bf16* __restrict__ W,     // [256][K]
    const float* __restrict__ bias,   // [256]
    __bf16* __restrict__ out,         // [M][256]
    int K, int relu)
{
  const int tid  = threadIdx.x;
  const int lane = tid & 31;
  const int wv   = tid >> 5;
  const int row0 = blockIdx.x * 128 + wv * 16;
  const int n0   = blockIdx.y * 64;
  const int rsub = (lane >> 4) << 3;
  const int nlan = lane & 15;

  v8f acc[4];
#pragma unroll
  for (int j = 0; j < 4; ++j)
#pragma unroll
    for (int v = 0; v < 8; ++v) acc[j][v] = 0.0f;

  const __bf16* arow = A + (size_t)row0 * K;
  for (int k0 = 0; k0 < K; k0 += 32) {
    v16bf a = frag_a(arow, K, k0, lane);
#pragma unroll
    for (int j = 0; j < 4; ++j)
      acc[j] = wmma_bf16(a, frag_b(W, K, n0 + 16 * j, k0, lane), acc[j]);
  }

#pragma unroll
  for (int j = 0; j < 4; ++j) {
    const int col = n0 + 16 * j + nlan;
    const float bb = bias[col];
#pragma unroll
    for (int v = 0; v < 8; ++v) {
      float val = acc[j][v] + bb;
      if (relu) val = fmaxf(val, 0.0f);
      out[(size_t)(row0 + rsub + v) * FCH + col] = (__bf16)val;
    }
  }
}

// ---------------------------------------------------------------------------
// FC3 (N=8 < WMMA tile): per-thread dot product + sigmoid*2-1, f32 out.
// ---------------------------------------------------------------------------
__global__ __launch_bounds__(256) void fc3_kernel(
    const __bf16* __restrict__ x,   // [B][256]
    const float* __restrict__ W3,   // [8][256]
    const float* __restrict__ b3,   // [8]
    float* __restrict__ out)        // [B][8]
{
  __shared__ float sW[OUTD * FCH];
  for (int i = threadIdx.x; i < OUTD * FCH; i += 256) sW[i] = W3[i];
  __syncthreads();

  const int idx = blockIdx.x * 256 + threadIdx.x;
  if (idx >= B_BATCH * OUTD) return;
  const int b = idx >> 3, o = idx & 7;
  float s = b3[o];
  const __bf16* xr = x + (size_t)b * FCH;
  const float* wr = sW + o * FCH;
#pragma unroll 8
  for (int k = 0; k < FCH; ++k) s += (float)xr[k] * wr[k];
  out[idx] = sigm(s) * 2.0f - 1.0f;   // sigmoid*(MAX-MIN)+MIN, MIN=-1, MAX=1
}

// ---------------------------------------------------------------------------
// Launcher
// ---------------------------------------------------------------------------
extern "C" void kernel_launch(void* const* d_in, const int* in_sizes, int n_in,
                              void* d_out, int out_size, void* d_ws, size_t ws_size,
                              hipStream_t stream) {
  (void)in_sizes; (void)n_in; (void)out_size; (void)ws_size;

  const float* obs  = (const float*)d_in[0];
  const float* Wih0 = (const float*)d_in[1];
  const float* Whh0 = (const float*)d_in[2];
  const float* bih0 = (const float*)d_in[3];
  const float* bhh0 = (const float*)d_in[4];
  const float* Wih1 = (const float*)d_in[5];
  const float* Whh1 = (const float*)d_in[6];
  const float* bih1 = (const float*)d_in[7];
  const float* bhh1 = (const float*)d_in[8];
  const float* W1   = (const float*)d_in[9];
  const float* b1   = (const float*)d_in[10];
  const float* W2   = (const float*)d_in[11];
  const float* b2   = (const float*)d_in[12];
  const float* W3   = (const float*)d_in[13];
  const float* b3   = (const float*)d_in[14];

  char* ws = (char*)d_ws;
  size_t off = 0;
  auto wsalloc = [&](size_t bytes) -> void* {
    void* p = (void*)(ws + off);
    off = (off + bytes + 255) & ~(size_t)255;
    return p;
  };

  __bf16* whh0_bf = (__bf16*)wsalloc((size_t)G4 * HID * 2);
  __bf16* wih1_bf = (__bf16*)wsalloc((size_t)G4 * HID * 2);
  __bf16* whh1_bf = (__bf16*)wsalloc((size_t)G4 * HID * 2);
  __bf16* w1_bf   = (__bf16*)wsalloc((size_t)FCH * T_SEQ * HID * 2);  // [256][8192]
  __bf16* w2_bf   = (__bf16*)wsalloc((size_t)FCH * FCH * 2);
  float*  bias0   = (float*)wsalloc((size_t)G4 * 4);
  float*  bias1   = (float*)wsalloc((size_t)G4 * 4);
  __bf16* x1      = (__bf16*)wsalloc((size_t)B_BATCH * T_SEQ * HID * 2);  // 64 MB
  __bf16* x2      = (__bf16*)wsalloc((size_t)B_BATCH * T_SEQ * HID * 2);  // 64 MB
  __bf16* out1    = (__bf16*)wsalloc((size_t)B_BATCH * FCH * 2);
  __bf16* out2    = (__bf16*)wsalloc((size_t)B_BATCH * FCH * 2);

  // Weight prep
  cvt_bf16_kernel<<<(G4 * HID + 255) / 256, 256, 0, stream>>>(Whh0, whh0_bf, G4 * HID);
  cvt_bf16_kernel<<<(G4 * HID + 255) / 256, 256, 0, stream>>>(Wih1, wih1_bf, G4 * HID);
  cvt_bf16_kernel<<<(G4 * HID + 255) / 256, 256, 0, stream>>>(Whh1, whh1_bf, G4 * HID);
  cvt_bf16_kernel<<<(FCH * T_SEQ * HID + 255) / 256, 256, 0, stream>>>(W1, w1_bf, FCH * T_SEQ * HID);
  cvt_bf16_kernel<<<(FCH * FCH + 255) / 256, 256, 0, stream>>>(W2, w2_bf, FCH * FCH);
  bias_add_kernel<<<1, 256, 0, stream>>>(bih0, bhh0, bias0, G4);
  bias_add_kernel<<<1, 256, 0, stream>>>(bih1, bhh1, bias1, G4);

  // Recurrent layers (wave-per-16-rows, c in VGPRs, h via LDS, WMMA GEMMs)
  lstm0_kernel<<<B_BATCH / 64, 128, 0, stream>>>(obs, Wih0, bias0, whh0_bf, x1);
  lstm1_kernel<<<B_BATCH / 64, 128, 0, stream>>>(x1, bias1, wih1_bf, whh1_bf, x2);

  // MLP head
  dim3 gfc(B_BATCH / 128, FCH / 64);
  fc_gemm_kernel<<<gfc, 256, 0, stream>>>(x2, w1_bf, b1, out1, T_SEQ * HID, 1);
  fc_gemm_kernel<<<gfc, 256, 0, stream>>>(out1, w2_bf, b2, out2, FCH, 1);
  fc3_kernel<<<(B_BATCH * OUTD + 255) / 256, 256, 0, stream>>>(out2, W3, b3, (float*)d_out);
}
